// LinearTemporalDiffusionTransformerDecoderLayer_32942399161117
// MI455X (gfx1250) — compile-verified, matrix operand order checked by reading, not run on hardware
//
#include <hip/hip_runtime.h>

// ---------------------------------------------------------------------------
// MI455X (gfx1250) implementation of the LinearTemporalDiffusionTransformer
// decoder layer. All dense GEMMs run through a bf16 WMMA tile kernel
// (v_wmma_f32_16x16x32_bf16, f32 accum). Weights are pre-transposed to
// bf16 [N,K] once per launch so BOTH the A and B tiles are contiguous and
// are DMA'd into double-buffered LDS by the Tensor Data Mover
// (tensor_load_to_lds, s_wait_tensorcnt(1) pipeline). Elementwise / softmax /
// tiny einsums stay in fp32 VALU (<1% of the FLOPs).
// ---------------------------------------------------------------------------

typedef __bf16 v16bf __attribute__((ext_vector_type(16)));
typedef float  v8f   __attribute__((ext_vector_type(8)));
typedef unsigned int u32x4 __attribute__((ext_vector_type(4)));
typedef int i32x4 __attribute__((ext_vector_type(4)));
typedef int i32x8 __attribute__((ext_vector_type(8)));

#define NEG_BIG (-1000000.0f)

#if defined(__HIP_DEVICE_COMPILE__) && __has_builtin(__builtin_amdgcn_tensor_load_to_lds)
#define HAVE_TDM 1
#else
#define HAVE_TDM 0
#endif

__device__ __forceinline__ unsigned short f2bf(float f) {
  unsigned u = __builtin_bit_cast(unsigned, f);
  u += 0x7FFFu + ((u >> 16) & 1u);           // round-to-nearest-even
  return (unsigned short)(u >> 16);
}

// Fragment loader (ISA 7.12.2): A (16x32) and B-from-[N,K] tiles share the
// same per-lane layout: lanes 0-15 kbase=0, lanes 16-31 kbase=8;
// vector elems 0..7 -> K=kbase+e, elems 8..15 -> K=kbase+8+e.
// LDS tile rows are padded to 40 halves.
__device__ __forceinline__ v16bf load_frag(const unsigned short* t, int rowStart, int lane) {
  int r  = rowStart + (lane & 15);
  int kb = (lane & 16) ? 8 : 0;
  union { v16bf v; u32x4 q[2]; } f;
  f.q[0] = *(const u32x4*)(t + r * 40 + kb);
  f.q[1] = *(const u32x4*)(t + r * 40 + kb + 16);
  return f.v;
}

#if HAVE_TDM
// D# (groups 0/1) for a 2D tile: 128 rows x 32 bf16, global row stride = K
// elements, LDS padding 4 dwords every 16 dwords -> LDS row stride 40 halves.
// Rows past rows_rem read out-of-tensor -> zero-filled by the TDM.
__device__ __forceinline__ void tdm_load_tile(unsigned lds_addr,
                                              const unsigned short* gptr,
                                              int K, int rows_rem) {
  unsigned long long ga = (unsigned long long)gptr;
  u32x4 g0;
  g0.x = 1u;                                     // count=1
  g0.y = lds_addr;                               // lds_addr
  g0.z = (unsigned)(ga & 0xFFFFFFFFull);         // global_addr[31:0]
  g0.w = (unsigned)((ga >> 32) & 0x01FFFFFFull) | (2u << 30);  // addr[56:32] | type=2
  unsigned td0 = (unsigned)K;
  unsigned td1 = (unsigned)(rows_rem < 0 ? 0 : rows_rem);
  i32x8 g1;
  g1[0] = (int)((1u << 16) | (1u << 20) | (3u << 22) | (3u << 25)); // 2B elems, pad 4dw/16dw
  g1[1] = (int)((td0 & 0xFFFFu) << 16);                            // tensor_dim0 lo
  g1[2] = (int)(((td0 >> 16) & 0xFFFFu) | ((td1 & 0xFFFFu) << 16));// dim0 hi | dim1 lo
  g1[3] = (int)(((td1 >> 16) & 0xFFFFu) | (32u << 16));            // dim1 hi | tile_dim0=32
  g1[4] = (int)(128u);                                             // tile_dim1=128, tile_dim2=0
  g1[5] = (int)td0;                                                // tensor_dim0_stride = K
  g1[6] = 0;
  g1[7] = 0;
  i32x4 z4 = {0, 0, 0, 0};
#if defined(__clang_major__) && (__clang_major__ >= 23)
  i32x8 z8 = {0, 0, 0, 0, 0, 0, 0, 0};
  __builtin_amdgcn_tensor_load_to_lds(g0, g1, z4, z4, z8, 0);
#else
  __builtin_amdgcn_tensor_load_to_lds(g0, g1, z4, z4, 0);
#endif
}
#endif

// ---------------------------------------------------------------------------
// GEMM: C[M,N] = A[M,K](bf16) @ W[K,N] + bias, with W given pre-transposed
// as Bt[N,K](bf16).  block = 256 threads (8 waves), tile 128x128, BK=32,
// double-buffered LDS; wave0/wave1 drive the TDM for the A/B tile of the
// NEXT k-step while all 8 waves compute the current one.
// wave (wm in 0..3, wn in 0..1) owns a 32x64 sub-tile = 2x4 WMMA tiles.
// mode 0: store fp32.  mode 1: store bf16(gelu_exact(x)).
// N and K are always multiples of 128/32 here; only M needs guarding.
// ---------------------------------------------------------------------------
__global__ void __launch_bounds__(256) gemm_bf16_wmma(
    const unsigned short* __restrict__ A,
    const unsigned short* __restrict__ Bt,
    const float* __restrict__ bias,
    float* __restrict__ Cf,
    unsigned short* __restrict__ Cbf,
    int M, int N, int K, int mode) {
  __shared__ unsigned short As[2][128 * 40];
  __shared__ unsigned short Bs[2][128 * 40];

  const int n0 = blockIdx.x * 128;
  const int m0 = blockIdx.y * 128;
  const int lane = threadIdx.x & 31;
  const int wave = threadIdx.x >> 5;
  const int wm = wave >> 1;
  const int wn = wave & 1;

  v8f zero = {0.f, 0.f, 0.f, 0.f, 0.f, 0.f, 0.f, 0.f};
  v8f acc[2][4];
#pragma unroll
  for (int i = 0; i < 2; ++i)
#pragma unroll
    for (int j = 0; j < 4; ++j) acc[i][j] = zero;

#if HAVE_TDM
  // Prologue: DMA tile 0 into buffer 0 (wave0 -> A, wave1 -> B).
  if (threadIdx.x < 32) {
    tdm_load_tile((unsigned)(unsigned long long)(void*)&As[0][0],
                  A + (size_t)m0 * K, K, M - m0);
  } else if (threadIdx.x < 64) {
    tdm_load_tile((unsigned)(unsigned long long)(void*)&Bs[0][0],
                  Bt + (size_t)n0 * K, K, N - n0);
  }
#endif

  int cur = 0;
  for (int k0 = 0; k0 < K; k0 += 32) {
    const bool hasNext = (k0 + 32 < K);
#if HAVE_TDM
    if (hasNext) {
      if (threadIdx.x < 32) {
        tdm_load_tile((unsigned)(unsigned long long)(void*)&As[cur ^ 1][0],
                      A + (size_t)m0 * K + (k0 + 32), K, M - m0);
      } else if (threadIdx.x < 64) {
        tdm_load_tile((unsigned)(unsigned long long)(void*)&Bs[cur ^ 1][0],
                      Bt + (size_t)n0 * K + (k0 + 32), K, N - n0);
      }
    }
    if (threadIdx.x < 64) {
      // per-wave in-order tensor counter: <=1 outstanding means the current
      // buffer's DMA has landed while the next one is still in flight.
      if (hasNext) __builtin_amdgcn_s_wait_tensorcnt(1);
      else         __builtin_amdgcn_s_wait_tensorcnt(0);
    }
    if (k0 + 64 < K)   // L2 warm-up two tiles ahead
      __builtin_prefetch(A + (size_t)(m0 + (threadIdx.x >> 1)) * K + k0 + 64, 0, 1);
#else
    {
      int r = threadIdx.x >> 1;
      int c = (threadIdx.x & 1) * 16;
      u32x4 a0 = {0, 0, 0, 0}, a1 = {0, 0, 0, 0};
      int gm = m0 + r;
      if (gm < M) {
        const u32x4* s = (const u32x4*)(A + (size_t)gm * K + k0 + c);
        a0 = s[0];
        a1 = s[1];
      }
      *(u32x4*)&As[cur][r * 40 + c] = a0;
      *(u32x4*)&As[cur][r * 40 + c + 8] = a1;
      const u32x4* s2 = (const u32x4*)(Bt + (size_t)(n0 + r) * K + k0 + c);
      *(u32x4*)&Bs[cur][r * 40 + c] = s2[0];
      *(u32x4*)&Bs[cur][r * 40 + c + 8] = s2[1];
    }
#endif
    __syncthreads();

    v16bf afr[2], bfr[4];
#pragma unroll
    for (int i = 0; i < 2; ++i) afr[i] = load_frag(&As[cur][0], wm * 32 + i * 16, lane);
#pragma unroll
    for (int j = 0; j < 4; ++j) bfr[j] = load_frag(&Bs[cur][0], wn * 64 + j * 16, lane);
#pragma unroll
    for (int i = 0; i < 2; ++i)
#pragma unroll
      for (int j = 0; j < 4; ++j)
        acc[i][j] = __builtin_amdgcn_wmma_f32_16x16x32_bf16(
            false, afr[i], false, bfr[j], (short)0, acc[i][j], false, false);
    __syncthreads();   // protect buffer (cur) before it is re-DMA'd next iter
    cur ^= 1;
  }

  // ---- epilogue: C layout (ISA 7.12.2): vgpr r -> M=r (+8 for lanes 16-31) ----
  const int mBase = m0 + wm * 32;
  const int nBase = n0 + wn * 64;
#pragma unroll
  for (int i = 0; i < 2; ++i) {
#pragma unroll
    for (int j = 0; j < 4; ++j) {
      int n = nBase + j * 16 + (lane & 15);
      int mr = mBase + i * 16 + ((lane & 16) ? 8 : 0);
      float bb = bias ? bias[n] : 0.0f;
#pragma unroll
      for (int r = 0; r < 8; ++r) {
        int m = mr + r;
        if (m < M) {
          float v = acc[i][j][r] + bb;
          if (mode == 0) {
            Cf[(size_t)m * N + n] = v;
          } else {
            float gl = 0.5f * v * (1.0f + erff(v * 0.70710678118654752f));
            Cbf[(size_t)m * N + n] = f2bf(gl);
          }
        }
      }
    }
  }
}

// ---------------------------------------------------------------------------
// Weight prep: fp32 W[K,N] -> bf16 Wt[N,K], tiled through LDS (once/launch).
// All weight dims are multiples of 32.
// ---------------------------------------------------------------------------
__global__ void transpose_bf16_kernel(const float* __restrict__ in,
                                      unsigned short* __restrict__ out,
                                      int Kd, int Nd) {
  __shared__ float tile[32][33];
  const int k0 = blockIdx.x * 32, n0 = blockIdx.y * 32;
  const int tx = threadIdx.x & 31, ty = threadIdx.x >> 5;  // 32x8
  for (int i = ty; i < 32; i += 8)
    tile[i][tx] = in[(size_t)(k0 + i) * Nd + n0 + tx];
  __syncthreads();
  for (int i = ty; i < 32; i += 8)
    out[(size_t)(n0 + i) * Kd + k0 + tx] = f2bf(tile[tx][i]);
}

// ---------------------------------------------------------------------------
// Support kernels (fp32 VALU; all with blockDim=256)
// ---------------------------------------------------------------------------
__global__ void ln_bf16_kernel(const float* __restrict__ X, const float* __restrict__ g,
                               const float* __restrict__ b, unsigned short* __restrict__ out) {
  const int row = blockIdx.x;
  const float* x = X + (size_t)row * 512;
  float s = 0.f, ss = 0.f, xv[2];
#pragma unroll
  for (int it = 0; it < 2; ++it) {
    int i = threadIdx.x + it * 256;
    xv[it] = x[i];
    s += xv[it];
    ss += xv[it] * xv[it];
  }
  __shared__ float rs[256], rq[256];
  rs[threadIdx.x] = s;
  rq[threadIdx.x] = ss;
  __syncthreads();
  for (int st = 128; st > 0; st >>= 1) {
    if (threadIdx.x < st) {
      rs[threadIdx.x] += rs[threadIdx.x + st];
      rq[threadIdx.x] += rq[threadIdx.x + st];
    }
    __syncthreads();
  }
  float mean = rs[0] * (1.0f / 512.0f);
  float var = rq[0] * (1.0f / 512.0f) - mean * mean;
  float inv = rsqrtf(var + 1e-5f);
#pragma unroll
  for (int it = 0; it < 2; ++it) {
    int i = threadIdx.x + it * 256;
    out[(size_t)row * 512 + i] = f2bf((xv[it] - mean) * inv * g[i] + b[i]);
  }
}

// ln(y)*(1+scale)+shift, then SiLU, to bf16.  E holds [scale(512) | shift(512)] per batch.
__global__ void styl_act_kernel(const float* __restrict__ Y, const float* __restrict__ g,
                                const float* __restrict__ b, const float* __restrict__ E,
                                unsigned short* __restrict__ out) {
  const int row = blockIdx.x;
  const int batch = row >> 10;  // T = 1024
  const float* x = Y + (size_t)row * 512;
  const float* e = E + (size_t)batch * 1024;
  float s = 0.f, ss = 0.f, xv[2];
#pragma unroll
  for (int it = 0; it < 2; ++it) {
    int i = threadIdx.x + it * 256;
    xv[it] = x[i];
    s += xv[it];
    ss += xv[it] * xv[it];
  }
  __shared__ float rs[256], rq[256];
  rs[threadIdx.x] = s;
  rq[threadIdx.x] = ss;
  __syncthreads();
  for (int st = 128; st > 0; st >>= 1) {
    if (threadIdx.x < st) {
      rs[threadIdx.x] += rs[threadIdx.x + st];
      rq[threadIdx.x] += rq[threadIdx.x + st];
    }
    __syncthreads();
  }
  float mean = rs[0] * (1.0f / 512.0f);
  float var = rq[0] * (1.0f / 512.0f) - mean * mean;
  float inv = rsqrtf(var + 1e-5f);
#pragma unroll
  for (int it = 0; it < 2; ++it) {
    int i = threadIdx.x + it * 256;
    float nl = (xv[it] - mean) * inv * g[i] + b[i];
    float h = nl * (1.0f + e[i]) + e[512 + i];
    out[(size_t)row * 512 + i] = f2bf(h / (1.0f + expf(-h)));
  }
}

__global__ void cvt_bf16_kernel(const float* __restrict__ in, unsigned short* __restrict__ out, int n) {
  int i = blockIdx.x * 256 + threadIdx.x;
  if (i < n) out[i] = f2bf(in[i]);
}

__global__ void silu_bf16_kernel(const float* __restrict__ in, unsigned short* __restrict__ out, int n) {
  int i = blockIdx.x * 256 + threadIdx.x;
  if (i >= n) return;
  float x = in[i];
  out[i] = f2bf(x / (1.0f + expf(-x)));
}

__global__ void swap_halves_bf16_kernel(const unsigned short* __restrict__ in,
                                        unsigned short* __restrict__ out, int half) {
  int i = blockIdx.x * 256 + threadIdx.x;
  int tot = half * 2;
  if (i >= tot) return;
  int j = i + half;
  if (j >= tot) j -= tot;
  out[i] = in[j];
}

__global__ void mask_k_kernel(float* __restrict__ Kf, const float* __restrict__ mask, int n) {
  int i = blockIdx.x * 256 + threadIdx.x;
  if (i >= n) return;
  Kf[i] += (1.0f - mask[i >> 9]) * NEG_BIG;
}

__global__ void mask_v_kernel(float* __restrict__ Vf, const float* __restrict__ mask, int n) {
  int i = blockIdx.x * 256 + threadIdx.x;
  if (i >= n) return;
  Vf[i] *= mask[i >> 9];
}

__global__ void softmax_q_kernel(float* __restrict__ Q, int rows) {
  int idx = blockIdx.x * 256 + threadIdx.x;
  if (idx >= rows * 8) return;
  float* p = Q + (size_t)(idx >> 3) * 512 + (size_t)(idx & 7) * 64;
  float mx = -3.4e38f;
  for (int i = 0; i < 64; ++i) mx = fmaxf(mx, p[i]);
  float s = 0.f;
  for (int i = 0; i < 64; ++i) {
    float e = expf(p[i] - mx);
    p[i] = e;
    s += e;
  }
  float inv = 1.0f / s;
  for (int i = 0; i < 64; ++i) p[i] *= inv;
}

__global__ void softmax_k_kernel(float* __restrict__ Kf, int Bn, int N) {
  int idx = blockIdx.x * 256 + threadIdx.x;
  if (idx >= Bn * 512) return;
  int b = idx >> 9, f = idx & 511;
  float* p = Kf + ((size_t)b * N) * 512 + f;
  float mx = -3.4e38f;
  for (int n = 0; n < N; ++n) mx = fmaxf(mx, p[(size_t)n * 512]);
  float s = 0.f;
  for (int n = 0; n < N; ++n) {
    float e = expf(p[(size_t)n * 512] - mx);
    p[(size_t)n * 512] = e;
    s += e;
  }
  float inv = 1.0f / s;
  for (int n = 0; n < N; ++n) p[(size_t)n * 512] *= inv;
}

// att[b,h,d,l] = sum_n k[b,n,h*64+d] * v[b,n,h*64+l]   (grid = 128 blocks of 256)
__global__ void kv_einsum_kernel(const float* __restrict__ Kf, const float* __restrict__ Vf,
                                 float* __restrict__ att, int N) {
  const int bh = blockIdx.x;
  const int b = bh >> 3, h = bh & 7;
  const int d = threadIdx.x >> 2;
  const int lb = (threadIdx.x & 3) * 16;
  float acc[16];
#pragma unroll
  for (int i = 0; i < 16; ++i) acc[i] = 0.f;
  const float* kp = Kf + (size_t)b * N * 512 + h * 64 + d;
  const float* vp = Vf + (size_t)b * N * 512 + h * 64 + lb;
  for (int n = 0; n < N; ++n) {
    float kv = kp[(size_t)n * 512];
    const float4* vr = (const float4*)(vp + (size_t)n * 512);
#pragma unroll
    for (int q4 = 0; q4 < 4; ++q4) {
      float4 v4 = vr[q4];
      acc[q4 * 4 + 0] = fmaf(kv, v4.x, acc[q4 * 4 + 0]);
      acc[q4 * 4 + 1] = fmaf(kv, v4.y, acc[q4 * 4 + 1]);
      acc[q4 * 4 + 2] = fmaf(kv, v4.z, acc[q4 * 4 + 2]);
      acc[q4 * 4 + 3] = fmaf(kv, v4.w, acc[q4 * 4 + 3]);
    }
  }
  float* op = att + (size_t)bh * 4096 + (size_t)d * 64 + lb;
#pragma unroll
  for (int i = 0; i < 16; ++i) op[i] = acc[i];
}

// y[b,t,h*64+l] = sum_d q[b,t,h*64+d] * att[b,h,d,l]  (grid (T/256, 8, Bn))
__global__ void qatt_kernel(const float* __restrict__ Q, const float* __restrict__ att,
                            float* __restrict__ Y, int T) {
  __shared__ float sA[4096];
  const int b = blockIdx.z, h = blockIdx.y;
  const float* ap = att + ((size_t)(b * 8 + h)) * 4096;
  for (int i = threadIdx.x; i < 4096; i += 256) sA[i] = ap[i];
  __syncthreads();
  const int t = blockIdx.x * 256 + threadIdx.x;
  const float* qp = Q + ((size_t)b * T + t) * 512 + h * 64;
  float qv[64];
#pragma unroll
  for (int i = 0; i < 64; ++i) qv[i] = qp[i];
  float* yp = Y + ((size_t)b * T + t) * 512 + h * 64;
  for (int l = 0; l < 64; ++l) {
    float a = 0.f;
#pragma unroll
    for (int d2 = 0; d2 < 64; ++d2) a = fmaf(qv[d2], sA[d2 * 64 + l], a);
    yp[l] = a;
  }
}

__global__ void add_inplace_kernel(float* __restrict__ X, const float* __restrict__ O, int n) {
  int i = blockIdx.x * 256 + threadIdx.x;
  if (i < n) X[i] += O[i];
}

__global__ void add_out_kernel(const float* __restrict__ X, const float* __restrict__ O,
                               float* __restrict__ out, int n) {
  int i = blockIdx.x * 256 + threadIdx.x;
  if (i < n) out[i] = X[i] + O[i];
}

// ---------------------------------------------------------------------------
// Host side
// ---------------------------------------------------------------------------
// Input order (JAX pytree flatten: dicts by sorted key, tuples in order):
enum {
  IN_X1 = 0, IN_X2, IN_XF, IN_EMB, IN_MASK, IN_TS,
  CA_K_W, CA_K_B, CA_LN_G, CA_LN_B, CA_Q_W, CA_Q_B,
  CA_SE_W, CA_SE_B, CA_SLNB, CA_SLNG, CA_SO_W, CA_SO_B,
  CA_TLN_G, CA_TLN_B, CA_V_W, CA_V_B,
  FF_L1_W, FF_L1_B, FF_L2_W, FF_L2_B,
  FF_SE_W, FF_SE_B, FF_SLNB, FF_SLNG, FF_SO_W, FF_SO_B,
  ICA_K_W, ICA_K_B, ICA_LN_G, ICA_LN_B, ICA_Q_W, ICA_Q_B,
  ICA_SE_W, ICA_SE_B, ICA_SLNB, ICA_SLNG, ICA_SO_W, ICA_SO_B,
  ICA_V_W, ICA_V_B,
  SA_K_W, SA_K_B, SA_LN_G, SA_LN_B, SA_Q_W, SA_Q_B,
  SA_SE_W, SA_SE_B, SA_SLNB, SA_SLNG, SA_SO_W, SA_SO_B,
  SA_V_W, SA_V_B,
  IN_COUNT
};

static constexpr size_t M2 = 16384, DD = 512, FF2 = 2048, MT = 1232;
static constexpr size_t AL(size_t x) { return (x + 255) & ~(size_t)255; }

static constexpr size_t OFF_X    = 0;
static constexpr size_t OFF_Q    = OFF_X    + AL(M2 * DD * 4);
static constexpr size_t OFF_K    = OFF_Q    + AL(M2 * DD * 4);
static constexpr size_t OFF_V    = OFF_K    + AL(M2 * DD * 4);
static constexpr size_t OFF_Y    = OFF_V    + AL(M2 * DD * 4);
static constexpr size_t OFF_O    = OFF_Y    + AL(M2 * DD * 4);
static constexpr size_t OFF_E    = OFF_O    + AL(M2 * DD * 4);
static constexpr size_t OFF_ATT  = OFF_E    + AL(16 * 1024 * 4);
static constexpr size_t OFF_NBF  = OFF_ATT  + AL(16 * 8 * 64 * 64 * 4);
static constexpr size_t OFF_NBF2 = OFF_NBF  + AL(M2 * DD * 2);
static constexpr size_t OFF_TNBF = OFF_NBF2 + AL(M2 * DD * 2);
static constexpr size_t OFF_SBF  = OFF_TNBF + AL(MT * DD * 2);
static constexpr size_t OFF_EBF  = OFF_SBF  + AL(M2 * DD * 2);
static constexpr size_t OFF_HBF  = OFF_EBF  + AL(16 * DD * 2);
static constexpr size_t OFF_WB   = OFF_HBF  + AL(M2 * FF2 * 2);

static inline void launch_gemm(const void* A, const void* Bt, const float* bias, void* Cf,
                               void* Cbf, int M, int N, int K, int mode, hipStream_t s) {
  dim3 grid(N / 128, (M + 127) / 128);
  gemm_bf16_wmma<<<grid, 256, 0, s>>>((const unsigned short*)A, (const unsigned short*)Bt,
                                      bias, (float*)Cf, (unsigned short*)Cbf, M, N, K, mode);
}

extern "C" void kernel_launch(void* const* d_in, const int* in_sizes, int n_in,
                              void* d_out, int out_size, void* d_ws, size_t ws_size,
                              hipStream_t stream) {
  (void)n_in; (void)out_size; (void)ws_size;
  char* ws = (char*)d_ws;
  float* X   = (float*)(ws + OFF_X);
  float* Qb  = (float*)(ws + OFF_Q);
  float* Kb  = (float*)(ws + OFF_K);
  float* Vb  = (float*)(ws + OFF_V);
  float* Yb  = (float*)(ws + OFF_Y);
  float* Ob  = (float*)(ws + OFF_O);
  float* Eb  = (float*)(ws + OFF_E);
  float* ATT = (float*)(ws + OFF_ATT);
  unsigned short* NBF  = (unsigned short*)(ws + OFF_NBF);
  unsigned short* NBF2 = (unsigned short*)(ws + OFF_NBF2);
  unsigned short* TNBF = (unsigned short*)(ws + OFF_TNBF);
  unsigned short* SBF  = (unsigned short*)(ws + OFF_SBF);
  unsigned short* EBF  = (unsigned short*)(ws + OFF_EBF);
  unsigned short* HBF  = (unsigned short*)(ws + OFF_HBF);
  unsigned short* WB   = (unsigned short*)(ws + OFF_WB);
  const float* maskp = (const float*)d_in[IN_MASK];
  const int NE = (int)((M2 * DD + 255) / 256);

  // --- weight prep: fp32 [K,N] -> bf16 [N,K] once per launch ---
  const unsigned short* wbf[IN_COUNT] = {nullptr};
  {
    struct WInfo { int idx, K, N; };
    const WInfo wi[19] = {
        {CA_K_W, 512, 512},  {CA_Q_W, 512, 512},  {CA_SE_W, 512, 1024},
        {CA_SO_W, 512, 512}, {CA_V_W, 512, 512},
        {FF_L1_W, 512, 2048},{FF_L2_W, 2048, 512},{FF_SE_W, 512, 1024},
        {FF_SO_W, 512, 512},
        {ICA_K_W, 512, 512}, {ICA_Q_W, 512, 512}, {ICA_SE_W, 512, 1024},
        {ICA_SO_W, 512, 512},{ICA_V_W, 512, 512},
        {SA_K_W, 512, 512},  {SA_Q_W, 512, 512},  {SA_SE_W, 512, 1024},
        {SA_SO_W, 512, 512}, {SA_V_W, 512, 512}};
    size_t off = 0;
    for (int i = 0; i < 19; ++i) {
      dim3 g(wi[i].K / 32, wi[i].N / 32);
      transpose_bf16_kernel<<<g, 256, 0, stream>>>((const float*)d_in[wi[i].idx],
                                                   WB + off, wi[i].K, wi[i].N);
      wbf[wi[i].idx] = WB + off;
      off = (off + (size_t)wi[i].K * wi[i].N + 127) & ~(size_t)127;
    }
  }

  auto run_styl = [&](int sew, int seb, int slnb, int slng, int sow, int sob, bool fin) {
    launch_gemm(EBF, wbf[sew], (const float*)d_in[seb], Eb, nullptr, 16, 1024, 512, 0, stream);
    styl_act_kernel<<<(int)M2, 256, 0, stream>>>(Yb, (const float*)d_in[slng],
                                                 (const float*)d_in[slnb], Eb, SBF);
    launch_gemm(SBF, wbf[sow], (const float*)d_in[sob], Ob, nullptr, (int)M2, 512, 512, 0, stream);
    if (fin)
      add_out_kernel<<<NE, 256, 0, stream>>>(X, Ob, (float*)d_out, (int)(M2 * DD));
    else
      add_inplace_kernel<<<NE, 256, 0, stream>>>(X, Ob, (int)(M2 * DD));
  };

  // x = cat([x1, x2])
  size_t halfElems = (size_t)in_sizes[IN_X1];
  hipMemcpyAsync(X, d_in[IN_X1], halfElems * 4, hipMemcpyDeviceToDevice, stream);
  hipMemcpyAsync(X + halfElems, d_in[IN_X2], halfElems * 4, hipMemcpyDeviceToDevice, stream);

  // silu(emb) -> bf16 (shared by all four styl blocks)
  silu_bf16_kernel<<<(16 * 512 + 255) / 256, 256, 0, stream>>>((const float*)d_in[IN_EMB], EBF, 16 * 512);

  // ---- self-attention ----
  ln_bf16_kernel<<<(int)M2, 256, 0, stream>>>(X, (const float*)d_in[SA_LN_G],
                                              (const float*)d_in[SA_LN_B], NBF);
  launch_gemm(NBF, wbf[SA_Q_W], (const float*)d_in[SA_Q_B], Qb, nullptr, (int)M2, 512, 512, 0, stream);
  launch_gemm(NBF, wbf[SA_K_W], (const float*)d_in[SA_K_B], Kb, nullptr, (int)M2, 512, 512, 0, stream);
  launch_gemm(NBF, wbf[SA_V_W], (const float*)d_in[SA_V_B], Vb, nullptr, (int)M2, 512, 512, 0, stream);
  mask_k_kernel<<<NE, 256, 0, stream>>>(Kb, maskp, (int)(M2 * DD));
  mask_v_kernel<<<NE, 256, 0, stream>>>(Vb, maskp, (int)(M2 * DD));
  softmax_q_kernel<<<(int)((M2 * 8 + 255) / 256), 256, 0, stream>>>(Qb, (int)M2);
  softmax_k_kernel<<<(16 * 512 + 255) / 256, 256, 0, stream>>>(Kb, 16, 1024);
  kv_einsum_kernel<<<128, 256, 0, stream>>>(Kb, Vb, ATT, 1024);
  qatt_kernel<<<dim3(4, 8, 16), 256, 0, stream>>>(Qb, ATT, Yb, 1024);
  run_styl(SA_SE_W, SA_SE_B, SA_SLNB, SA_SLNG, SA_SO_W, SA_SO_B, false);

  // ---- text cross-attention ----
  ln_bf16_kernel<<<(int)M2, 256, 0, stream>>>(X, (const float*)d_in[CA_LN_G],
                                              (const float*)d_in[CA_LN_B], NBF);
  ln_bf16_kernel<<<(int)MT, 256, 0, stream>>>((const float*)d_in[IN_XF],
                                              (const float*)d_in[CA_TLN_G],
                                              (const float*)d_in[CA_TLN_B], TNBF);
  launch_gemm(NBF, wbf[CA_Q_W], (const float*)d_in[CA_Q_B], Qb, nullptr, (int)M2, 512, 512, 0, stream);
  launch_gemm(TNBF, wbf[CA_K_W], (const float*)d_in[CA_K_B], Kb, nullptr, (int)MT, 512, 512, 0, stream);
  launch_gemm(TNBF, wbf[CA_V_W], (const float*)d_in[CA_V_B], Vb, nullptr, (int)MT, 512, 512, 0, stream);
  softmax_q_kernel<<<(int)((M2 * 8 + 255) / 256), 256, 0, stream>>>(Qb, (int)M2);
  softmax_k_kernel<<<(16 * 512 + 255) / 256, 256, 0, stream>>>(Kb, 16, 77);
  kv_einsum_kernel<<<128, 256, 0, stream>>>(Kb, Vb, ATT, 77);
  qatt_kernel<<<dim3(4, 8, 16), 256, 0, stream>>>(Qb, ATT, Yb, 1024);
  run_styl(CA_SE_W, CA_SE_B, CA_SLNB, CA_SLNG, CA_SO_W, CA_SO_B, false);

  // ---- interaction cross-attention (K/V from swapped halves) ----
  ln_bf16_kernel<<<(int)M2, 256, 0, stream>>>(X, (const float*)d_in[ICA_LN_G],
                                              (const float*)d_in[ICA_LN_B], NBF);
  swap_halves_bf16_kernel<<<NE, 256, 0, stream>>>(NBF, NBF2, (int)(M2 * DD / 2));
  launch_gemm(NBF, wbf[ICA_Q_W], (const float*)d_in[ICA_Q_B], Qb, nullptr, (int)M2, 512, 512, 0, stream);
  launch_gemm(NBF2, wbf[ICA_K_W], (const float*)d_in[ICA_K_B], Kb, nullptr, (int)M2, 512, 512, 0, stream);
  launch_gemm(NBF2, wbf[ICA_V_W], (const float*)d_in[ICA_V_B], Vb, nullptr, (int)M2, 512, 512, 0, stream);
  mask_k_kernel<<<NE, 256, 0, stream>>>(Kb, maskp, (int)(M2 * DD));
  softmax_q_kernel<<<(int)((M2 * 8 + 255) / 256), 256, 0, stream>>>(Qb, (int)M2);
  softmax_k_kernel<<<(16 * 512 + 255) / 256, 256, 0, stream>>>(Kb, 16, 1024);
  kv_einsum_kernel<<<128, 256, 0, stream>>>(Kb, Vb, ATT, 1024);
  qatt_kernel<<<dim3(4, 8, 16), 256, 0, stream>>>(Qb, ATT, Yb, 1024);
  run_styl(ICA_SE_W, ICA_SE_B, ICA_SLNB, ICA_SLNG, ICA_SO_W, ICA_SO_B, false);

  // ---- FFN (gelu fused into GEMM epilogue, bf16 hidden) ----
  cvt_bf16_kernel<<<NE, 256, 0, stream>>>(X, NBF, (int)(M2 * DD));
  launch_gemm(NBF, wbf[FF_L1_W], (const float*)d_in[FF_L1_B], nullptr, HBF, (int)M2, 2048, 512, 1, stream);
  launch_gemm(HBF, wbf[FF_L2_W], (const float*)d_in[FF_L2_B], Yb, nullptr, (int)M2, 512, 2048, 0, stream);
  run_styl(FF_SE_W, FF_SE_B, FF_SLNB, FF_SLNG, FF_SO_W, FF_SO_B, true);  // writes d_out
}
